// MultiheadAttention_47536698032374
// MI455X (gfx1250) — matrix operand order repaired and checked by reading, not compiled, not run on hardware
//
#include <hip/hip_runtime.h>
#include <hip/hip_bf16.h>

typedef __attribute__((ext_vector_type(16))) __bf16 v16bf;
typedef __attribute__((ext_vector_type(8)))  __bf16 v8bf;
typedef __attribute__((ext_vector_type(8)))  float  v8f;

constexpr int NDIM  = 1024;
constexpr int SEQ   = 2048;
constexpr int BATCH = 2;
constexpr int HEADS = 16;
constexpr int DH    = 64;

// ---- fragment builders (bf16 only; data pre-packed) ------------------------
// A fragment 16x32 bf16, contiguous k order chosen here (B matches): lane group g (lane>>4):
//   element e -> K = g*16 + e          (loads [g*16 .. g*16+15])
static __device__ inline v16bf fragA_bf(const unsigned short* rowk, int g) {
    const unsigned short* p = rowk + g * 16;
    v8bf lo = *(const v8bf*)(p);
    v8bf hi = *(const v8bf*)(p + 8);
    v16bf r;
#pragma unroll
    for (int i = 0; i < 8; ++i) { r[i] = lo[i]; r[i + 8] = hi[i]; }
    return r;
}
// B fragment 32x16 bf16, same contiguous k order as A here: lane group g: element e -> K = g*16 + e
static __device__ inline v16bf fragB_bf(const unsigned short* rowk, int g) {
    const unsigned short* p = rowk + g * 16;
    v8bf lo = *(const v8bf*)(p);
    v8bf hi = *(const v8bf*)(p + 8);
    v16bf r;
#pragma unroll
    for (int i = 0; i < 8; ++i) { r[i] = lo[i]; r[i + 8] = hi[i]; }
    return r;
}
static __device__ inline unsigned short f2bf(float f) {
    __bf16 b = (__bf16)f;
    return __builtin_bit_cast(unsigned short, b);
}

#define WMMA_BF16(A, B, C) \
    __builtin_amdgcn_wmma_f32_16x16x32_bf16(false, (A), false, (B), (short)0, (C), false, false)

// ---- kernel 0: fp32 -> bf16 pack (one-shot; removes cvt from GEMM loops) ---
__global__ void __launch_bounds__(256) pack_bf16_kernel(
    const float* __restrict__ src, unsigned short* __restrict__ dst) {
    const int i = (blockIdx.x * 256 + threadIdx.x) * 8;
    const float4 f0 = *(const float4*)(src + i);
    const float4 f1 = *(const float4*)(src + i + 4);
    v8bf v;
    v[0] = (__bf16)f0.x; v[1] = (__bf16)f0.y; v[2] = (__bf16)f0.z; v[3] = (__bf16)f0.w;
    v[4] = (__bf16)f1.x; v[5] = (__bf16)f1.y; v[6] = (__bf16)f1.z; v[7] = (__bf16)f1.w;
    *(v8bf*)(dst + i) = v;
}

// ---- kernel 1: fused QKV projection  y = x @ W^T ---------------------------
// One wave computes a 16x64 output tile: 1 A-fragment feeds 4 WMMA per chunk.
// Q,K stored bf16 [B,H,S,Dh]; V stored bf16 transposed [B,H,Dh,S].
__global__ void __launch_bounds__(256) qkv_proj_kernel(
    const unsigned short* __restrict__ X,
    const unsigned short* __restrict__ Wq, const unsigned short* __restrict__ Wk,
    const unsigned short* __restrict__ Wv,
    unsigned short* __restrict__ Qws, unsigned short* __restrict__ Kws,
    unsigned short* __restrict__ Vt) {
    const int lane = threadIdx.x & 31;
    const int g = lane >> 4, ln = lane & 15;
    const int wave = blockIdx.x * 8 + (threadIdx.x >> 5);
    const int proj = wave / (256 * 16);        // 0=Q 1=K 2=V
    const int rem  = wave % (256 * 16);
    const int m0 = (rem >> 4) * 16;            // row tile  [0,4096)
    const int n0 = (rem & 15) * 64;            // col tile  [0,1024), 64 wide
    const unsigned short* W = (proj == 0) ? Wq : (proj == 1) ? Wk : Wv;

    const unsigned short* pa = X + (m0 + ln) * NDIM;
    const unsigned short* pb0 = W + (n0 + ln) * NDIM;
    const unsigned short* pb1 = pb0 + 16 * NDIM;
    const unsigned short* pb2 = pb0 + 32 * NDIM;
    const unsigned short* pb3 = pb0 + 48 * NDIM;

    v8f acc0 = {}, acc1 = {}, acc2 = {}, acc3 = {};
#pragma unroll 4
    for (int k0 = 0; k0 < NDIM; k0 += 32) {
        const v16bf a = fragA_bf(pa + k0, g);
        acc0 = WMMA_BF16(a, fragB_bf(pb0 + k0, g), acc0);
        acc1 = WMMA_BF16(a, fragB_bf(pb1 + k0, g), acc1);
        acc2 = WMMA_BF16(a, fragB_bf(pb2 + k0, g), acc2);
        acc3 = WMMA_BF16(a, fragB_bf(pb3 + k0, g), acc3);
    }

    const int h = n0 >> 6;                     // whole 64-wide tile = one head
#pragma unroll
    for (int f = 0; f < 4; ++f) {
        const v8f acc = (f == 0) ? acc0 : (f == 1) ? acc1 : (f == 2) ? acc2 : acc3;
        const int dh = f * 16 + ln;
#pragma unroll
        for (int r = 0; r < 8; ++r) {
            const int m  = m0 + r + g * 8;
            const int bb = m >> 11;            // batch
            const int s  = m & 2047;           // seq pos
            const unsigned short bits = f2bf(acc[r]);
            if (proj == 0)      Qws[((bb * HEADS + h) * SEQ + s) * DH + dh] = bits;
            else if (proj == 1) Kws[((bb * HEADS + h) * SEQ + s) * DH + dh] = bits;
            else                Vt[((bb * HEADS + h) * DH + dh) * SEQ + s]  = bits;
        }
    }
}

// ---- kernel 2: flash attention (one wave = 32 q rows of one head) ----------
// K/V B-fragments loaded once per 32-key block, reused by both 16-row q tiles.
__global__ void __launch_bounds__(256) attn_kernel(
    const unsigned short* __restrict__ Q, const unsigned short* __restrict__ K,
    const unsigned short* __restrict__ Vt, unsigned short* __restrict__ Y) {
    __shared__ unsigned short pLds[8][2 * 16 * 32];   // per-wave P staging (2KB)

    const int lane = threadIdx.x & 31;
    const int g = lane >> 4, ln = lane & 15;
    const int wv = threadIdx.x >> 5;
    const int wave = blockIdx.x * 8 + wv;
    const int bh = wave >> 6;                 // (b*H + h)
    const int s0 = (wave & 63) * 32;
    const int bb = bh >> 4, h = bh & 15;

    // Q A-fragments: 2 q-tiles x 2 Dh chunks, kept resident
    v16bf aq[2][2];
#pragma unroll
    for (int t = 0; t < 2; ++t) {
        const unsigned short* qrow = Q + (bh * SEQ + s0 + t * 16 + ln) * DH;
        aq[t][0] = fragA_bf(qrow, g);
        aq[t][1] = fragA_bf(qrow + 32, g);
    }

    float rowmax[2][8], rowsum[2][8];
    v8f o[2][4];
#pragma unroll
    for (int t = 0; t < 2; ++t)
#pragma unroll
        for (int r = 0; r < 8; ++r) { rowmax[t][r] = -1e30f; rowsum[t][r] = 0.0f; }
#pragma unroll
    for (int t = 0; t < 2; ++t)
#pragma unroll
        for (int f = 0; f < 4; ++f) o[t][f] = (v8f){};

    unsigned short* myP = pLds[wv];
    const float scale = 0.125f;               // 1/sqrt(64)

    for (int j2 = 0; j2 < SEQ / 32; ++j2) {
        // K fragments for 32 keys (2 key subtiles x 2 Dh chunks) — loaded once
        const unsigned short* kr0 = K + (bh * SEQ + j2 * 32 + ln) * DH;
        const unsigned short* kr1 = kr0 + 16 * DH;
        const v16bf bk00 = fragB_bf(kr0, g),      bk01 = fragB_bf(kr0 + 32, g);
        const v16bf bk10 = fragB_bf(kr1, g),      bk11 = fragB_bf(kr1 + 32, g);
        // V fragments (32 keys x 64 dh) — loaded once
        const unsigned short* vbp = Vt + (bh * DH + ln) * SEQ + j2 * 32;
        v16bf bv[4];
        bv[0] = fragB_bf(vbp, g);
        bv[1] = fragB_bf(vbp + 16 * SEQ, g);
        bv[2] = fragB_bf(vbp + 32 * SEQ, g);
        bv[3] = fragB_bf(vbp + 48 * SEQ, g);

#pragma unroll
        for (int t = 0; t < 2; ++t) {
            v8f c0 = {}, c1 = {};
            c0 = WMMA_BF16(aq[t][0], bk00, c0);
            c0 = WMMA_BF16(aq[t][1], bk01, c0);
            c1 = WMMA_BF16(aq[t][0], bk10, c1);
            c1 = WMMA_BF16(aq[t][1], bk11, c1);

            // online softmax over this 32-key block
#pragma unroll
            for (int r = 0; r < 8; ++r) {
                const float s0v = c0[r] * scale;
                const float s1v = c1[r] * scale;
                float mx = fmaxf(s0v, s1v);
                mx = fmaxf(mx, __shfl_xor(mx, 1, 32));
                mx = fmaxf(mx, __shfl_xor(mx, 2, 32));
                mx = fmaxf(mx, __shfl_xor(mx, 4, 32));
                mx = fmaxf(mx, __shfl_xor(mx, 8, 32));
                const float nm   = fmaxf(rowmax[t][r], mx);
                const float corr = __expf(rowmax[t][r] - nm);
                rowmax[t][r] = nm;
                const float p0 = __expf(s0v - nm);
                const float p1 = __expf(s1v - nm);
                float ts = p0 + p1;
                ts += __shfl_xor(ts, 1, 32);
                ts += __shfl_xor(ts, 2, 32);
                ts += __shfl_xor(ts, 4, 32);
                ts += __shfl_xor(ts, 8, 32);
                rowsum[t][r] = rowsum[t][r] * corr + ts;
                o[t][0][r] *= corr; o[t][1][r] *= corr;
                o[t][2][r] *= corr; o[t][3][r] *= corr;
                // stage probs (C layout -> LDS row-major 16x32)
                const int m = r + g * 8;
                myP[t * 512 + m * 32 + ln]      = f2bf(p0);
                myP[t * 512 + m * 32 + 16 + ln] = f2bf(p1);
            }
            // transpose C-layout -> A-fragment via wave-private LDS (DS in-order)
            const v16bf ap = fragA_bf(myP + t * 512 + ln * 32, g);
            o[t][0] = WMMA_BF16(ap, bv[0], o[t][0]);
            o[t][1] = WMMA_BF16(ap, bv[1], o[t][1]);
            o[t][2] = WMMA_BF16(ap, bv[2], o[t][2]);
            o[t][3] = WMMA_BF16(ap, bv[3], o[t][3]);
        }
    }

    // normalize and emit Y in [B,S,D] bf16 for the output projection
#pragma unroll
    for (int t = 0; t < 2; ++t)
#pragma unroll
        for (int r = 0; r < 8; ++r) {
            const float inv = 1.0f / rowsum[t][r];
            const int m = s0 + t * 16 + r + g * 8;
            unsigned short* yp = Y + (bb * SEQ + m) * NDIM + h * DH + ln;
            yp[0]  = f2bf(o[t][0][r] * inv);
            yp[16] = f2bf(o[t][1][r] * inv);
            yp[32] = f2bf(o[t][2][r] * inv);
            yp[48] = f2bf(o[t][3][r] * inv);
        }
}

// ---- kernel 3: output projection  out = Y @ wo^T + bo ----------------------
__global__ void __launch_bounds__(256) oproj_kernel(
    const unsigned short* __restrict__ Y, const unsigned short* __restrict__ Wo,
    const float* __restrict__ bo, float* __restrict__ out) {
    const int lane = threadIdx.x & 31;
    const int g = lane >> 4, ln = lane & 15;
    const int wave = blockIdx.x * 8 + (threadIdx.x >> 5);
    const int m0 = (wave >> 4) * 16;
    const int n0 = (wave & 15) * 64;

    const unsigned short* pa  = Y + (m0 + ln) * NDIM;
    const unsigned short* pb0 = Wo + (n0 + ln) * NDIM;
    const unsigned short* pb1 = pb0 + 16 * NDIM;
    const unsigned short* pb2 = pb0 + 32 * NDIM;
    const unsigned short* pb3 = pb0 + 48 * NDIM;

    v8f acc0 = {}, acc1 = {}, acc2 = {}, acc3 = {};
#pragma unroll 4
    for (int k0 = 0; k0 < NDIM; k0 += 32) {
        const v16bf a = fragA_bf(pa + k0, g);
        acc0 = WMMA_BF16(a, fragB_bf(pb0 + k0, g), acc0);
        acc1 = WMMA_BF16(a, fragB_bf(pb1 + k0, g), acc1);
        acc2 = WMMA_BF16(a, fragB_bf(pb2 + k0, g), acc2);
        acc3 = WMMA_BF16(a, fragB_bf(pb3 + k0, g), acc3);
    }
#pragma unroll
    for (int f = 0; f < 4; ++f) {
        const v8f acc = (f == 0) ? acc0 : (f == 1) ? acc1 : (f == 2) ? acc2 : acc3;
        const float bias = bo[n0 + f * 16 + ln];
#pragma unroll
        for (int r = 0; r < 8; ++r) {
            const int m = m0 + r + g * 8;
            out[m * NDIM + n0 + f * 16 + ln] = acc[r] + bias;
        }
    }
}

// ---- host launch -----------------------------------------------------------
extern "C" void kernel_launch(void* const* d_in, const int* in_sizes, int n_in,
                              void* d_out, int out_size, void* d_ws, size_t ws_size,
                              hipStream_t stream) {
    const float* x  = (const float*)d_in[0];
    const float* wq = (const float*)d_in[1];
    const float* wk = (const float*)d_in[2];
    const float* wv = (const float*)d_in[3];
    const float* wo = (const float*)d_in[4];
    const float* bo = (const float*)d_in[5];
    float* out = (float*)d_out;

    constexpr size_t XN  = (size_t)BATCH * SEQ * NDIM;   // 4,194,304
    constexpr size_t WN  = (size_t)NDIM * NDIM;          // 1,048,576
    constexpr size_t SEG = (size_t)BATCH * HEADS * SEQ * DH;

    unsigned short* Xbf = (unsigned short*)d_ws;
    unsigned short* Wqb = Xbf + XN;
    unsigned short* Wkb = Wqb + WN;
    unsigned short* Wvb = Wkb + WN;
    unsigned short* Wob = Wvb + WN;
    unsigned short* Qws = Wob + WN;
    unsigned short* Kws = Qws + SEG;
    unsigned short* Vt  = Kws + SEG;
    unsigned short* Ybf = Vt + SEG;   // total ~48MB

    // one-shot bf16 packing (8 elems/thread)
    pack_bf16_kernel<<<XN / 2048, 256, 0, stream>>>(x, Xbf);
    pack_bf16_kernel<<<WN / 2048, 256, 0, stream>>>(wq, Wqb);
    pack_bf16_kernel<<<WN / 2048, 256, 0, stream>>>(wk, Wkb);
    pack_bf16_kernel<<<WN / 2048, 256, 0, stream>>>(wv, Wvb);
    pack_bf16_kernel<<<WN / 2048, 256, 0, stream>>>(wo, Wob);

    // 3 projections x 256 row-tiles x 16 col-tiles(64w) = 12288 waves
    qkv_proj_kernel<<<1536, 256, 0, stream>>>(Xbf, Wqb, Wkb, Wvb, Qws, Kws, Vt);
    // 32 (b,h) x 64 q-tiles(32 rows) = 2048 waves
    attn_kernel<<<256, 256, 0, stream>>>(Qws, Kws, Vt, Ybf);
    // 256 x 16 tiles = 4096 waves
    oproj_kernel<<<512, 256, 0, stream>>>(Ybf, Wob, bo, out);
}
